// Real_Gated_Linear_Recurrent_Unit_74131135529624
// MI455X (gfx1250) — matrix operand order repaired and checked by reading, not compiled, not run on hardware
//
#include <hip/hip_runtime.h>
#include <hip/hip_bf16.h>

typedef __attribute__((ext_vector_type(16))) __bf16 v16bf;
typedef __attribute__((ext_vector_type(8)))  __bf16 v8bf;
typedef __attribute__((ext_vector_type(4)))  __bf16 v4bf;
typedef __attribute__((ext_vector_type(8)))  float  v8f;

#define BB 4
#define SS 4096
#define DD 2048
#define MM (BB*SS)   // 16384 rows (B*S)
#define BM 128       // block tile M
#define BN 64        // block tile N
#define KT 32        // K tile (one bf16 WMMA step)
#define KP 40        // padded LDS K pitch (elements): 80B rows, 16B aligned, bank-conflict free
#define KITERS (DD / KT)

__device__ __forceinline__ v4bf cvt4(float4 f) {
    v4bf r;
    r[0] = (__bf16)f.x; r[1] = (__bf16)f.y;
    r[2] = (__bf16)f.z; r[3] = (__bf16)f.w;
    return r;
}

__device__ __forceinline__ v16bf join16(v8bf lo, v8bf hi) {
    return __builtin_shufflevector(lo, hi, 0,1,2,3,4,5,6,7,8,9,10,11,12,13,14,15);
}

__device__ __forceinline__ float sigmoidf_(float z) {
    return 1.0f / (1.0f + __expf(-z));
}

// ---------------------------------------------------------------------------
// Kernel 1: fused gate GEMMs (r = sigma(x Wa^T), i = sigma(x Wx^T)) + RG-LRU
// elementwise. Writes a -> Aout (workspace), u -> Uout (= d_out).
// Grid: (DD/BN, MM/BM), 256 threads = 8 waves (4 along M x 2 along N).
// Ping-pong double-buffered LDS: one barrier per K step; global loads for
// tile k+1 overlap the 8 WMMAs on tile k.
// ---------------------------------------------------------------------------
__global__ __launch_bounds__(256) void rglru_gates_wmma(
    const float* __restrict__ x,  const float* __restrict__ Wa,
    const float* __restrict__ Wx, const float* __restrict__ ba,
    const float* __restrict__ bx, const float* __restrict__ Lam,
    float* __restrict__ Aout, float* __restrict__ Uout)
{
    __shared__ __align__(16) __bf16 sX [2][BM * KP];
    __shared__ __align__(16) __bf16 sWa[2][BN * KP];
    __shared__ __align__(16) __bf16 sWx[2][BN * KP];

    const int tid  = threadIdx.x;
    const int lane = tid & 31;
    const int wid  = tid >> 5;
    const int wm   = wid & 3;             // wave coord along M (0..3) -> 32 rows each
    const int wn   = wid >> 2;            // wave coord along N (0..1) -> 32 cols each
    const int m0   = blockIdx.y * BM;
    const int n0   = blockIdx.x * BN;

    // cooperative-load geometry (per thread): X = 4 float4, W = 2 float4 each
    const int xrow[4] = { (0*256+tid) >> 3, (1*256+tid) >> 3, (2*256+tid) >> 3, (3*256+tid) >> 3 };
    const int wrow[2] = { (0*256+tid) >> 3, (1*256+tid) >> 3 };
    const int c4      = (tid & 7) * 4;

    v8f accR[2][2], accI[2][2];
    #pragma unroll
    for (int mt = 0; mt < 2; ++mt)
        #pragma unroll
        for (int nt = 0; nt < 2; ++nt) { accR[mt][nt] = {}; accI[mt][nt] = {}; }

    const int arow = lane & 15;           // fragment row (M for A, N for B)
    const int akb  = (lane >> 4) * 16;    // A fragment K base (0 or 16)
    const int bko  = (lane >> 4) * 16;    // B fragment K base (0 or 16)

    float4 rx[4], rwa[2], rwx[2];

    // ---- prologue: stage tile 0 into buffer 0
    #pragma unroll
    for (int r = 0; r < 4; ++r)
        rx[r] = *reinterpret_cast<const float4*>(x + (size_t)(m0 + xrow[r]) * DD + c4);
    #pragma unroll
    for (int r = 0; r < 2; ++r) {
        rwa[r] = *reinterpret_cast<const float4*>(Wa + (size_t)(n0 + wrow[r]) * DD + c4);
        rwx[r] = *reinterpret_cast<const float4*>(Wx + (size_t)(n0 + wrow[r]) * DD + c4);
    }
    #pragma unroll
    for (int r = 0; r < 4; ++r)
        *reinterpret_cast<v4bf*>(&sX[0][xrow[r] * KP + c4]) = cvt4(rx[r]);
    #pragma unroll
    for (int r = 0; r < 2; ++r) {
        *reinterpret_cast<v4bf*>(&sWa[0][wrow[r] * KP + c4]) = cvt4(rwa[r]);
        *reinterpret_cast<v4bf*>(&sWx[0][wrow[r] * KP + c4]) = cvt4(rwx[r]);
    }
    __syncthreads();

    for (int kk = 0; kk < KITERS; ++kk) {
        const int cur = kk & 1;
        const int k1  = (kk + 1) * KT;

        // ---- issue global loads for tile kk+1 (overlap with WMMAs below)
        if (k1 < DD) {
            #pragma unroll
            for (int r = 0; r < 4; ++r)
                rx[r] = *reinterpret_cast<const float4*>(x + (size_t)(m0 + xrow[r]) * DD + k1 + c4);
            #pragma unroll
            for (int r = 0; r < 2; ++r) {
                rwa[r] = *reinterpret_cast<const float4*>(Wa + (size_t)(n0 + wrow[r]) * DD + k1 + c4);
                rwx[r] = *reinterpret_cast<const float4*>(Wx + (size_t)(n0 + wrow[r]) * DD + k1 + c4);
            }
            // hint tile kk+2 toward the caches (global_prefetch_b8)
            __builtin_prefetch(x  + (size_t)(m0 + xrow[0]) * DD + k1 + KT, 0, 0);
            __builtin_prefetch(Wa + (size_t)(n0 + wrow[0]) * DD + k1 + KT, 0, 0);
            __builtin_prefetch(Wx + (size_t)(n0 + wrow[0]) * DD + k1 + KT, 0, 0);
        }

        // ---- fragments from buffer `cur` (contiguous K per half-wave; same K order for A and B)
        // A 16x32 bf16: lane<16 -> M=lane, K in {akb..akb+15} (contiguous)
        v16bf af[2];
        #pragma unroll
        for (int mt = 0; mt < 2; ++mt) {
            const __bf16* pa = &sX[cur][(wm * 32 + mt * 16 + arow) * KP + akb];
            af[mt] = join16(*reinterpret_cast<const v8bf*>(pa),
                            *reinterpret_cast<const v8bf*>(pa + 8));
        }
        // B 32x16 bf16: lane -> N col; half-wave selects K 0..15 / 16..31 (contiguous)
        v16bf bfa[2], bfx[2];
        #pragma unroll
        for (int nt = 0; nt < 2; ++nt) {
            const __bf16* p1 = &sWa[cur][(wn * 32 + nt * 16 + arow) * KP + bko];
            bfa[nt] = join16(*reinterpret_cast<const v8bf*>(p1),
                             *reinterpret_cast<const v8bf*>(p1 + 8));
            const __bf16* p2 = &sWx[cur][(wn * 32 + nt * 16 + arow) * KP + bko];
            bfx[nt] = join16(*reinterpret_cast<const v8bf*>(p2),
                             *reinterpret_cast<const v8bf*>(p2 + 8));
        }

        // ---- 8 WMMAs: shared A fragments feed both gate GEMMs
        #pragma unroll
        for (int mt = 0; mt < 2; ++mt)
            #pragma unroll
            for (int nt = 0; nt < 2; ++nt) {
                accR[mt][nt] = __builtin_amdgcn_wmma_f32_16x16x32_bf16(
                    false, af[mt], false, bfa[nt], (short)0, accR[mt][nt], false, false);
                accI[mt][nt] = __builtin_amdgcn_wmma_f32_16x16x32_bf16(
                    false, af[mt], false, bfx[nt], (short)0, accI[mt][nt], false, false);
            }

        // ---- store staged tile kk+1 into the other buffer (no reader conflict:
        // all waves are between barrier kk-1 and kk, reading only buffer `cur`)
        if (k1 < DD) {
            const int nxt = cur ^ 1;
            #pragma unroll
            for (int r = 0; r < 4; ++r)
                *reinterpret_cast<v4bf*>(&sX[nxt][xrow[r] * KP + c4]) = cvt4(rx[r]);
            #pragma unroll
            for (int r = 0; r < 2; ++r) {
                *reinterpret_cast<v4bf*>(&sWa[nxt][wrow[r] * KP + c4]) = cvt4(rwa[r]);
                *reinterpret_cast<v4bf*>(&sWx[nxt][wrow[r] * KP + c4]) = cvt4(rwx[r]);
            }
        }
        __syncthreads();
    }

    // ---- fused RG-LRU epilogue
    // C/D layout: VGPR v, lane l -> M = v + (l>=16 ? 8 : 0), N = l & 15
    const int mh = (lane >> 4) * 8;
    #pragma unroll
    for (int nt = 0; nt < 2; ++nt) {
        const int n_g  = n0 + wn * 32 + nt * 16 + (lane & 15);
        const float lam  = Lam[n_g];
        const float bav  = ba[n_g];
        const float bxv  = bx[n_g];
        // coef = -C * softplus(-Lambda), C = 8 ; -Lambda < 0 so log1p form is stable
        const float coef = -8.0f * __logf(1.0f + __expf(-lam));
        #pragma unroll
        for (int mt = 0; mt < 2; ++mt) {
            #pragma unroll
            for (int v = 0; v < 8; ++v) {
                const int m_g = m0 + wm * 32 + mt * 16 + v + mh;
                const size_t off = (size_t)m_g * DD + n_g;
                const float rg = sigmoidf_(accR[mt][nt][v] + bav);
                const float ig = sigmoidf_(accI[mt][nt][v] + bxv);
                const float a  = __expf(coef * rg);
                const float xv = x[off];
                const float u  = sqrtf(fmaxf(1.0f - a * a, 0.0f)) * ig * xv;
                Aout[off] = a;
                Uout[off] = u;
            }
        }
    }
}

// ---------------------------------------------------------------------------
// Kernel 2: diagonal linear recurrence h_t = a_t * h_{t-1} + u_t, in place
// over Y (u on entry, y on exit). One thread per (b,d) channel. Only 256
// waves exist -> latency-bound, so unroll x8: 16 coalesced loads in flight
// per thread; the serial dependency is confined to the 8 FMAs.
// ---------------------------------------------------------------------------
__global__ __launch_bounds__(256) void rglru_scan(
    const float* __restrict__ A, float* __restrict__ Y)
{
    const int c = blockIdx.x * blockDim.x + threadIdx.x;   // 0 .. B*D-1
    const int b = c / DD;
    const int d = c - b * DD;
    const float* pa = A + (size_t)b * SS * DD + d;
    float*       py = Y + (size_t)b * SS * DD + d;

    float h = 0.0f;
    for (int t = 0; t < SS; t += 8) {
        float av[8], uv[8];
        #pragma unroll
        for (int i = 0; i < 8; ++i) {
            av[i] = pa[(size_t)(t + i) * DD];
            uv[i] = py[(size_t)(t + i) * DD];
        }
        #pragma unroll
        for (int i = 0; i < 8; ++i) {
            h = fmaf(av[i], h, uv[i]);
            py[(size_t)(t + i) * DD] = h;
        }
    }
}

extern "C" void kernel_launch(void* const* d_in, const int* in_sizes, int n_in,
                              void* d_out, int out_size, void* d_ws, size_t ws_size,
                              hipStream_t stream) {
    (void)in_sizes; (void)n_in; (void)out_size; (void)ws_size;
    const float* x   = (const float*)d_in[0];
    const float* Wa  = (const float*)d_in[1];
    const float* Wx  = (const float*)d_in[2];
    const float* ba  = (const float*)d_in[3];
    const float* bx  = (const float*)d_in[4];
    const float* Lam = (const float*)d_in[5];
    float* y   = (float*)d_out;
    float* Aws = (float*)d_ws;    // B*S*D floats = 128 MiB for the decay gate a

    dim3 grid(DD / BN, MM / BM);  // (32, 128)
    rglru_gates_wmma<<<grid, 256, 0, stream>>>(x, Wa, Wx, ba, bx, Lam, Aws, y);

    rglru_scan<<<(BB * DD) / 256, 256, 0, stream>>>(Aws, y);
}
